// ISDT_19095424598413
// MI455X (gfx1250) — compile-verified, matrix operand order checked by reading, not run on hardware
//
#include <hip/hip_runtime.h>
#include <hip/hip_bf16.h>

typedef __bf16 bf16;
typedef __attribute__((ext_vector_type(16))) __bf16 v16bf;
typedef __attribute__((ext_vector_type(8)))  float  v8f;
typedef unsigned int v4u  __attribute__((ext_vector_type(4)));
typedef int          v8i_ __attribute__((ext_vector_type(8)));
typedef int          v4i_ __attribute__((ext_vector_type(4)));

#define N_TOK 32768
#define D_IN  2048
#define D_H   1024
#define CS    2048
#define TOPM  256
#define LDSTR 72   // 64 elems + 4-dword pad = 144B row stride (16B aligned, bank-conflict-free)

#if defined(__HIP_DEVICE_COMPILE__) && __has_builtin(__builtin_amdgcn_tensor_load_to_lds) && __has_builtin(__builtin_amdgcn_s_wait_tensorcnt)
#define HAVE_TDM 1
#else
#define HAVE_TDM 0
#endif

// ---------------- workspace layout (bytes) ----------------
static const size_t OFF_H0B = 0;                          // 128MB (dead after GEMM1)
static const size_t OFF_Z   = 0;                          // 64MB  (reuse)
static const size_t OFF_CN  = 64ull  * 1024 * 1024;       // 4MB   (reuse)
static const size_t OFF_W1T = 128ull * 1024 * 1024;       // 4MB
static const size_t OFF_W2T = 132ull * 1024 * 1024;       // 2MB
static const size_t OFF_WMT = 134ull * 1024 * 1024;       // 2MB
static const size_t OFF_WTT = 136ull * 1024 * 1024;       // 2MB
static const size_t OFF_WPT = 138ull * 1024 * 1024;       // 2MB
static const size_t OFF_H1  = 140ull * 1024 * 1024;       // 64MB (dead after GEMM2)
static const size_t OFF_S   = OFF_H1;                     // 128KB (reuse H1)
static const size_t OFF_H   = 204ull * 1024 * 1024;       // 64MB

// ---------------- TDM 2D tile load: global -> LDS, padded rows ----------------
#if HAVE_TDM
__device__ __forceinline__ void tdm_load_2d(unsigned lds_addr, const bf16* gptr,
                                            unsigned tile_d0, unsigned tile_d1,
                                            unsigned tensor_d0, unsigned tensor_d1,
                                            unsigned long stride0_elems) {
  unsigned long ga = (unsigned long)gptr;
  v4u g0;
  g0[0] = 1u;                                            // count=1 (valid user D#)
  g0[1] = lds_addr;                                      // LDS byte address
  g0[2] = (unsigned)(ga & 0xffffffffu);                  // global_addr[31:0]
  g0[3] = (unsigned)((ga >> 32) & 0x1ffffffu) | (2u << 30); // global_addr[56:32], type=2
  v8i_ g1;
  g1[0] = (int)((1u << 16)      // data_size = 2 bytes
              | (1u << 20)      // pad_enable
              | (4u << 22)      // pad_interval: 32 dwords (=128B = one 64-elem row)
              | (3u << 25));    // pad_amount: 4 dwords (=16B = 8 elems)
  g1[1] = (int)((tensor_d0 & 0xffffu) << 16);            // tensor_dim0[15:0]
  g1[2] = (int)(((tensor_d0 >> 16) & 0xffffu) | ((tensor_d1 & 0xffffu) << 16));
  g1[3] = (int)(((tensor_d1 >> 16) & 0xffffu) | ((tile_d0 & 0xffffu) << 16));
  g1[4] = (int)(tile_d1 & 0xffffu);                      // tile_dim1; tile_dim2=0
  g1[5] = (int)(stride0_elems & 0xffffffffu);            // tensor_dim0_stride[31:0]
  g1[6] = (int)((stride0_elems >> 32) & 0xffffu);        // stride0[47:32]; stride1=0
  g1[7] = 0;
  v4i_ z4 = {0, 0, 0, 0};
  v8i_ z8 = {0, 0, 0, 0, 0, 0, 0, 0};
  __builtin_amdgcn_tensor_load_to_lds(g0, g1, z4, z4, z8, 0);
}
#endif

// ---------------- f32 -> bf16 convert, 8 elems/thread ----------------
__global__ __launch_bounds__(256) void k_cvt_bf16(const float* __restrict__ in,
                                                  bf16* __restrict__ out, long n) {
  long i = ((long)blockIdx.x * blockDim.x + threadIdx.x) * 8;
  if (i + 8 > n) return;
  float4 a = *(const float4*)(in + i);
  float4 b = *(const float4*)(in + i + 4);
  union { bf16 h[8]; uint4 q; } u;
  u.h[0]=(bf16)a.x; u.h[1]=(bf16)a.y; u.h[2]=(bf16)a.z; u.h[3]=(bf16)a.w;
  u.h[4]=(bf16)b.x; u.h[5]=(bf16)b.y; u.h[6]=(bf16)b.z; u.h[7]=(bf16)b.w;
  *(uint4*)(out + i) = u.q;
}

// ---------------- transpose f32 [R x C] -> bf16 [C x R] ----------------
__global__ __launch_bounds__(256) void k_transpose_cvt(const float* __restrict__ in,
                                                       bf16* __restrict__ out,
                                                       int R, int C) {
  __shared__ float tile[32][33];
  int x = blockIdx.x * 32 + threadIdx.x;
  int y0 = blockIdx.y * 32;
  for (int i = threadIdx.y; i < 32; i += 8)
    tile[i][threadIdx.x] = in[(size_t)(y0 + i) * C + x];
  __syncthreads();
  int ox = y0 + threadIdx.x;
  int oy0 = blockIdx.x * 32;
  for (int i = threadIdx.y; i < 32; i += 8)
    out[(size_t)(oy0 + i) * R + ox] = (bf16)tile[threadIdx.x][i];
}

// ---------------- row-normalize codebook f32 -> bf16 ----------------
__global__ __launch_bounds__(256) void k_norm_rows(const float* __restrict__ Cb,
                                                   bf16* __restrict__ out, int cols) {
  __shared__ float red[256];
  int r = blockIdx.x;
  const float* src = Cb + (size_t)r * cols;
  float ss = 0.f;
  for (int k = threadIdx.x; k < cols; k += 256) { float v = src[k]; ss += v * v; }
  red[threadIdx.x] = ss; __syncthreads();
  for (int s = 128; s > 0; s >>= 1) {
    if (threadIdx.x < s) red[threadIdx.x] += red[threadIdx.x + s];
    __syncthreads();
  }
  float inv = rsqrtf(red[0] + 1e-12f);
  for (int k = threadIdx.x; k < cols; k += 256)
    out[(size_t)r * cols + k] = (bf16)(src[k] * inv);
}

// ---------------- bf16 WMMA GEMM: Out[M,N] = act(A[M,K] * Bt[N,K]^T + bias) ----------------
// BM=128, BN=64, BK=64; TDM double-buffered LDS (padded rows); 8 waves, wave = 32x32
template <int ACT>
__global__ __launch_bounds__(256) void k_gemm(const bf16* __restrict__ A,
                                              const bf16* __restrict__ Bt,
                                              const float* __restrict__ bias,
                                              bf16* __restrict__ Out,
                                              int M, int Nn, int K) {
  __shared__ bf16 sA[2][128 * LDSTR];   // 36KB
  __shared__ bf16 sB[2][64 * LDSTR];    // 18KB
  const int tid  = threadIdx.x;
  const int lane = tid & 31;
  const int wave = tid >> 5;
  const int wr = wave >> 1;                  // 0..3 -> 32-row slab
  const int wc = wave & 1;                   // 0..1 -> 32-col slab
  const int rowBlk = blockIdx.y * 128;
  const int colBlk = blockIdx.x * 64;

  v8f zero = {};
  v8f acc[2][2];
  #pragma unroll
  for (int i = 0; i < 2; i++)
    #pragma unroll
    for (int j = 0; j < 2; j++) acc[i][j] = zero;

  const bf16* Ablk = A  + (size_t)rowBlk * K;
  const bf16* Bblk = Bt + (size_t)colBlk * K;
  const int nstage = K >> 6;

#if HAVE_TDM
  if (wave == 0) {
    tdm_load_2d((unsigned)(size_t)(&sA[0][0]), Ablk, 64, 128, (unsigned)K, (unsigned)M, (unsigned long)K);
    tdm_load_2d((unsigned)(size_t)(&sB[0][0]), Bblk, 64, 64,  (unsigned)K, (unsigned)Nn, (unsigned long)K);
  }
#else
  { // fallback cooperative load of stage 0, padded layout
    int r = tid >> 3, c8 = (tid & 7) * 8;
    for (int p = 0; p < 4; p++)
      *(uint4*)(&sA[0][(r + p * 32) * LDSTR + c8]) = *(const uint4*)(&Ablk[(size_t)(r + p * 32) * K + c8]);
    for (int p = 0; p < 2; p++)
      *(uint4*)(&sB[0][(r + p * 32) * LDSTR + c8]) = *(const uint4*)(&Bblk[(size_t)(r + p * 32) * K + c8]);
  }
#endif

  for (int s = 0; s < nstage; s++) {
    const int buf = s & 1;
#if HAVE_TDM
    if (wave == 0) __builtin_amdgcn_s_wait_tensorcnt(0);  // stage s tiles landed in LDS
#endif
    __syncthreads();
    if (s + 1 < nstage) {
      const int nk = (s + 1) << 6;
#if HAVE_TDM
      if (wave == 0) {
        tdm_load_2d((unsigned)(size_t)(&sA[buf ^ 1][0]), Ablk + nk, 64, 128, (unsigned)K, (unsigned)M, (unsigned long)K);
        tdm_load_2d((unsigned)(size_t)(&sB[buf ^ 1][0]), Bblk + nk, 64, 64,  (unsigned)K, (unsigned)Nn, (unsigned long)K);
      }
#else
      int r = tid >> 3, c8 = (tid & 7) * 8;
      for (int p = 0; p < 4; p++)
        *(uint4*)(&sA[buf ^ 1][(r + p * 32) * LDSTR + c8]) = *(const uint4*)(&Ablk[(size_t)(r + p * 32) * K + nk + c8]);
      for (int p = 0; p < 2; p++)
        *(uint4*)(&sB[buf ^ 1][(r + p * 32) * LDSTR + c8]) = *(const uint4*)(&Bblk[(size_t)(r + p * 32) * K + nk + c8]);
#endif
    }
    // compute on buf
    #pragma unroll
    for (int kk = 0; kk < 64; kk += 32) {
      const int koff = kk + (lane >> 4) * 8;     // lanes 0-15: +0 ; lanes 16-31: +8
      v16bf a[2], b[2];
      #pragma unroll
      for (int i = 0; i < 2; i++) {
        const bf16* p0 = &sA[buf][(wr * 32 + i * 16 + (lane & 15)) * LDSTR + koff];
        ((uint4*)&a[i])[0] = *(const uint4*)(p0);        // K = base .. base+7
        ((uint4*)&a[i])[1] = *(const uint4*)(p0 + 16);   // K = base+16 .. base+23
      }
      #pragma unroll
      for (int j = 0; j < 2; j++) {
        const bf16* p0 = &sB[buf][(wc * 32 + j * 16 + (lane & 15)) * LDSTR + koff];
        ((uint4*)&b[j])[0] = *(const uint4*)(p0);
        ((uint4*)&b[j])[1] = *(const uint4*)(p0 + 16);
      }
      #pragma unroll
      for (int i = 0; i < 2; i++)
        #pragma unroll
        for (int j = 0; j < 2; j++)
          acc[i][j] = __builtin_amdgcn_wmma_f32_16x16x32_bf16(
              false, a[i], false, b[j], (short)0, acc[i][j], false, false);
    }
    __syncthreads();   // all reads of buf done before stage s+2 overwrites it
  }

  // epilogue: D layout -> lane<16: M=i, lane>=16: M=i+8 ; N = lane&15
  const int rbase = rowBlk + wr * 32 + ((lane >> 4) ? 8 : 0);
  const int cbase = colBlk + wc * 32 + (lane & 15);
  #pragma unroll
  for (int i = 0; i < 2; i++) {
    #pragma unroll
    for (int j = 0; j < 2; j++) {
      int c = cbase + j * 16;
      float bv = bias[c];
      #pragma unroll
      for (int e = 0; e < 8; e++) {
        int r = rbase + i * 16 + e;
        float v = acc[i][j][e] + bv;
        if (ACT) v = fmaxf(v, 0.f);
        Out[(size_t)r * Nn + c] = (bf16)v;
      }
    }
  }
}

// ---------------- fused VQ: argmax_c ( z . Cn[c] ), Cn row-normalized ----------------
__global__ __launch_bounds__(256) void k_vq(const bf16* __restrict__ Z,
                                            const bf16* __restrict__ Cn,
                                            int* __restrict__ codes, int mod) {
  __shared__ bf16 sA[16 * D_H];              // 32KB
  __shared__ float sBV[8][16];
  __shared__ int   sBI[8][16];
  const int tid  = threadIdx.x;
  const int lane = tid & 31;
  const int wave = tid >> 5;
  const int rowBlk = blockIdx.x * 16;

  for (int i = 0; i < 8; i++) {
    int c = tid + 256 * i;
    int r = c >> 7, c8 = (c & 127) * 8;
    *(uint4*)(&sA[r * D_H + c8]) = *(const uint4*)(&Z[(size_t)(rowBlk + r) * D_H + c8]);
  }
  __syncthreads();

  float bv[8]; int bi[8];
  #pragma unroll
  for (int e = 0; e < 8; e++) { bv[e] = -INFINITY; bi[e] = 0; }

  const int koffBase = (lane >> 4) * 8;
  for (int ct = 0; ct < 16; ct++) {
    const int c0 = wave * 256 + ct * 16;
    v8f acc = {};
    for (int kk = 0; kk < D_H; kk += 32) {
      const int koff = kk + koffBase;
      v16bf a, b;
      const bf16* pa = &sA[(lane & 15) * D_H + koff];
      ((uint4*)&a)[0] = *(const uint4*)(pa);
      ((uint4*)&a)[1] = *(const uint4*)(pa + 16);
      const bf16* pb = &Cn[(size_t)(c0 + (lane & 15)) * D_H + koff];
      ((uint4*)&b)[0] = *(const uint4*)(pb);
      ((uint4*)&b)[1] = *(const uint4*)(pb + 16);
      acc = __builtin_amdgcn_wmma_f32_16x16x32_bf16(false, a, false, b,
                                                    (short)0, acc, false, false);
    }
    const int myCode = c0 + (lane & 15);
    #pragma unroll
    for (int e = 0; e < 8; e++) {
      float v = acc[e];
      if (v > bv[e] || (v == bv[e] && myCode < bi[e])) { bv[e] = v; bi[e] = myCode; }
    }
  }

  #pragma unroll
  for (int e = 0; e < 8; e++) {
    for (int m = 1; m < 16; m <<= 1) {
      float ov = __shfl_xor(bv[e], m);
      int   oi = __shfl_xor(bi[e], m);
      if (ov > bv[e] || (ov == bv[e] && oi < bi[e])) { bv[e] = ov; bi[e] = oi; }
    }
  }
  if ((lane & 15) == 0) {
    int half = lane >> 4;
    #pragma unroll
    for (int e = 0; e < 8; e++) { sBV[wave][half * 8 + e] = bv[e]; sBI[wave][half * 8 + e] = bi[e]; }
  }
  __syncthreads();
  if (tid < 16) {
    float best = -INFINITY; int idx = 0;
    for (int w = 0; w < 8; w++) {
      float v = sBV[w][tid]; int ii = sBI[w][tid];
      if (v > best || (v == best && ii < idx)) { best = v; idx = ii; }
    }
    codes[(size_t)(rowBlk + tid) * 3 + mod] = idx;
  }
}

// ---------------- key score ----------------
__global__ __launch_bounds__(256) void k_score(const bf16* __restrict__ H,
                                               const float* __restrict__ ksw,
                                               const float* __restrict__ ksb,
                                               float* __restrict__ s) {
  int row  = blockIdx.x * 8 + (threadIdx.x >> 5);
  int lane = threadIdx.x & 31;
  const bf16* h = H + (size_t)row * D_H;
  float acc = 0.f;
  for (int k = lane; k < D_H; k += 32) acc += (float)h[k] * ksw[k];
  for (int m = 16; m > 0; m >>= 1) acc += __shfl_xor(acc, m);
  if (lane == 0) s[row] = 1.f / (1.f + __expf(-(acc + ksb[0])));
}

// ---------------- top-256 select + gather (single block) ----------------
__global__ __launch_bounds__(1024) void k_topk(float* __restrict__ s,
                                               const int* __restrict__ codes,
                                               int* __restrict__ out_key,
                                               int* __restrict__ out_gath, int n) {
  __shared__ float sv[1024];
  __shared__ int   si[1024];
  int t = threadIdx.x;
  for (int m = 0; m < TOPM; m++) {
    float bvv = -INFINITY; int bii = 0x7fffffff;
    for (int k = t; k < n; k += 1024) {
      float v = s[k];
      if (v > bvv || (v == bvv && k < bii)) { bvv = v; bii = k; }
    }
    sv[t] = bvv; si[t] = bii; __syncthreads();
    for (int st = 512; st > 0; st >>= 1) {
      if (t < st) {
        float ov = sv[t + st]; int oi = si[t + st];
        if (ov > sv[t] || (ov == sv[t] && oi < si[t])) { sv[t] = ov; si[t] = oi; }
      }
      __syncthreads();
    }
    if (t == 0) {
      int idx = si[0];
      out_key[m] = idx;
      out_gath[m * 3 + 0] = codes[(size_t)idx * 3 + 0];
      out_gath[m * 3 + 1] = codes[(size_t)idx * 3 + 1];
      out_gath[m * 3 + 2] = codes[(size_t)idx * 3 + 2];
      s[idx] = -INFINITY;
    }
    __syncthreads();
  }
}

extern "C" void kernel_launch(void* const* d_in, const int* in_sizes, int n_in,
                              void* d_out, int out_size, void* d_ws, size_t ws_size,
                              hipStream_t stream) {
  (void)in_sizes; (void)n_in; (void)out_size; (void)ws_size;
  const float* h0  = (const float*)d_in[0];
  const float* W1  = (const float*)d_in[1];
  const float* b1  = (const float*)d_in[2];
  const float* W2  = (const float*)d_in[3];
  const float* b2  = (const float*)d_in[4];
  const float* Wm  = (const float*)d_in[5];
  const float* bm  = (const float*)d_in[6];
  const float* Wt  = (const float*)d_in[7];
  const float* bt  = (const float*)d_in[8];
  const float* Wp  = (const float*)d_in[9];
  const float* bp  = (const float*)d_in[10];
  const float* Cm  = (const float*)d_in[11];
  const float* Ct  = (const float*)d_in[12];
  const float* Cp  = (const float*)d_in[13];
  const float* ksw = (const float*)d_in[14];
  const float* ksb = (const float*)d_in[15];

  char* ws = (char*)d_ws;
  bf16* h0b = (bf16*)(ws + OFF_H0B);
  bf16* W1T = (bf16*)(ws + OFF_W1T);
  bf16* W2T = (bf16*)(ws + OFF_W2T);
  bf16* WmT = (bf16*)(ws + OFF_WMT);
  bf16* WtT = (bf16*)(ws + OFF_WTT);
  bf16* WpT = (bf16*)(ws + OFF_WPT);
  bf16* H1  = (bf16*)(ws + OFF_H1);
  bf16* Hb  = (bf16*)(ws + OFF_H);
  bf16* Zb  = (bf16*)(ws + OFF_Z);
  bf16* Cn  = (bf16*)(ws + OFF_CN);
  float* sc = (float*)(ws + OFF_S);

  int* codes = (int*)d_out;                       // [32768,3]
  int* key   = codes + (size_t)N_TOK * 3;         // [256]
  int* gath  = key + TOPM;                        // [256,3]

  {
    long cnt = (long)N_TOK * D_IN;
    k_cvt_bf16<<<(unsigned)(cnt / 8 / 256), 256, 0, stream>>>(h0, h0b, cnt);
  }
  k_transpose_cvt<<<dim3(D_H / 32, D_IN / 32), dim3(32, 8), 0, stream>>>(W1, W1T, D_IN, D_H);
  k_transpose_cvt<<<dim3(D_H / 32, D_H / 32),  dim3(32, 8), 0, stream>>>(W2, W2T, D_H, D_H);
  k_transpose_cvt<<<dim3(D_H / 32, D_H / 32),  dim3(32, 8), 0, stream>>>(Wm, WmT, D_H, D_H);
  k_transpose_cvt<<<dim3(D_H / 32, D_H / 32),  dim3(32, 8), 0, stream>>>(Wt, WtT, D_H, D_H);
  k_transpose_cvt<<<dim3(D_H / 32, D_H / 32),  dim3(32, 8), 0, stream>>>(Wp, WpT, D_H, D_H);

  k_gemm<1><<<dim3(D_H / 64, N_TOK / 128), 256, 0, stream>>>(h0b, W1T, b1, H1, N_TOK, D_H, D_IN);
  k_gemm<1><<<dim3(D_H / 64, N_TOK / 128), 256, 0, stream>>>(H1,  W2T, b2, Hb, N_TOK, D_H, D_H);

  k_score<<<N_TOK / 8, 256, 0, stream>>>(Hb, ksw, ksb, sc);

  const bf16*  WT[3] = { WmT, WtT, WpT };
  const float* bb[3] = { bm, bt, bp };
  const float* CB[3] = { Cm, Ct, Cp };
  for (int mod = 0; mod < 3; mod++) {
    k_norm_rows<<<CS, 256, 0, stream>>>(CB[mod], Cn, D_H);
    k_gemm<0><<<dim3(D_H / 64, N_TOK / 128), 256, 0, stream>>>(Hb, WT[mod], bb[mod], Zb,
                                                               N_TOK, D_H, D_H);
    k_vq<<<N_TOK / 16, 256, 0, stream>>>(Zb, Cn, codes, mod);
  }
  k_topk<<<1, 1024, 0, stream>>>(sc, codes, key, gath, N_TOK);
}
